// GeneratorSDF_90623809945983
// MI455X (gfx1250) — matrix operand order repaired and checked by reading, not compiled, not run on hardware
//
#include <hip/hip_runtime.h>
#include <hip/hip_bf16.h>

typedef __attribute__((ext_vector_type(16))) _Float16 v16h;
typedef __attribute__((ext_vector_type(8)))  _Float16 v8h;
typedef __attribute__((ext_vector_type(2)))  _Float16 h2;
typedef __attribute__((ext_vector_type(8)))  float    v8f;

// ---- geometry / net constants -------------------------------------------
#define GRIDD   128
#define NPTS    (GRIDD * GRIDD * GRIDD)      // 2,097,152
#define CIN1    67
#define K1PAD   96                            // cin of layer1 padded to 3 K-chunks of 32
#define SW1     104                           // LDS row stride (halfs) for W1  (208B = 16*13)
#define SW2     136                           // 272B = 16*17
#define SW3     72                            // 144B = 16*9
#define SW4     40                            // 80B  = 16*5
#define SACT    136                           // activation row stride (halfs), 272B
#define BLOCKS  4096
#define ITERS   4                             // tiles-of-128-points per block

// ---- WMMA wrapper --------------------------------------------------------
__device__ __forceinline__ v8f wmma16(v16h a, v16h b, v8f c) {
    return __builtin_amdgcn_wmma_f32_16x16x32_f16(false, a, false, b,
                                                  (short)0, c, false, false);
}

// A fragment: 16x32 f16 tile of row-major weights in LDS.
// lanes 0-15: M = mbase+lane, K = kbase+{0..7, 16..23}
// lanes16-31: M = mbase+lane-16, K = kbase+{8..15, 24..31}
__device__ __forceinline__ v16h load_A(const _Float16* w, int mbase, int kbase,
                                       int stride, int lane) {
    const int m = mbase + (lane & 15);
    const int g = (lane >> 4) * 8;
    const _Float16* p = w + m * stride + kbase + g;
    v8h lo = *(const v8h*)(p);        // K = kbase+g   .. +7
    v8h hi = *(const v8h*)(p + 16);   // K = kbase+g+16 .. +23
    v16h a;
#pragma unroll
    for (int i = 0; i < 8; ++i) { a[i] = lo[i]; a[i + 8] = hi[i]; }
    return a;
}

// B fragment: 32x16 f16, activations stored [point][channel] with row stride.
// lanes 0-15: N = lane,    K = kbase+{0..7, 16..23}
// lanes16-31: N = lane-16, K = kbase+{8..15, 24..31}
__device__ __forceinline__ v16h load_B(const _Float16* act, int kbase,
                                       int stride, int lane) {
    const _Float16* p = act + (lane & 15) * stride + kbase + (lane >> 4) * 8;
    v8h lo = *(const v8h*)(p);
    v8h hi = *(const v8h*)(p + 16);
    v16h b;
#pragma unroll
    for (int i = 0; i < 8; ++i) { b[i] = lo[i]; b[i + 8] = hi[i]; }
    return b;
}

__device__ __forceinline__ void store_pair(_Float16* dst, float v0, float v1) {
    h2 t; t[0] = (_Float16)v0; t[1] = (_Float16)v1;
    *(h2*)dst = t;                    // 4B-aligned packed f16 pair -> ds_store_b32
}

__global__ __launch_bounds__(256)
void sdf_mlp_wmma(const float* __restrict__ x,
                  const float* __restrict__ W1, const float* __restrict__ b1,
                  const float* __restrict__ W2, const float* __restrict__ b2,
                  const float* __restrict__ W3, const float* __restrict__ b3,
                  const float* __restrict__ W4, const float* __restrict__ b4,
                  float* __restrict__ out) {
    __shared__ __align__(16) _Float16 sW1[128 * SW1];
    __shared__ __align__(16) _Float16 sW2[64 * SW2];
    __shared__ __align__(16) _Float16 sW3[32 * SW3];
    __shared__ __align__(16) _Float16 sW4[16 * SW4];
    __shared__ __align__(16) _Float16 sX[64];
    __shared__ __align__(16) _Float16 sAct[8][16 * SACT];
    __shared__ float sB1[128], sB2[64], sB3[32], sB4;

    const int tid = threadIdx.x;

    // ---- stage + convert weights/biases/latent into LDS (f16) -----------
    for (int i = tid; i < 128 * SW1; i += 256) {
        int r = i / SW1, c = i - r * SW1;
        float v = (c < CIN1) ? W1[r * CIN1 + c] : 0.0f;
        sW1[i] = (_Float16)v;
    }
    for (int i = tid; i < 64 * SW2; i += 256) {
        int r = i / SW2, c = i - r * SW2;
        float v = (c < 128) ? W2[r * 128 + c] : 0.0f;
        sW2[i] = (_Float16)v;
    }
    for (int i = tid; i < 32 * SW3; i += 256) {
        int r = i / SW3, c = i - r * SW3;
        float v = (c < 64) ? W3[r * 64 + c] : 0.0f;
        sW3[i] = (_Float16)v;
    }
    for (int i = tid; i < 16 * SW4; i += 256) {
        int r = i / SW4, c = i - r * SW4;
        float v = (r == 0 && c < 32) ? W4[c] : 0.0f;
        sW4[i] = (_Float16)v;
    }
    if (tid < 128) sB1[tid] = b1[tid];
    if (tid < 64)  { sB2[tid] = b2[tid]; sX[tid] = (_Float16)x[tid]; }
    if (tid < 32)  sB3[tid] = b3[tid];
    if (tid == 0)  sB4 = b4[0];
    __syncthreads();

    const int wave = tid >> 5;
    const int lane = tid & 31;
    const int col  = lane & 15;      // point column within tile
    const int grp  = lane >> 4;      // half-wave group
    _Float16* actw = &sAct[wave][0];

    for (int it = 0; it < ITERS; ++it) {
        const int tile  = (blockIdx.x * ITERS + it) * 8 + wave;
        const int pbase = tile * 16;
        const int pidx  = pbase + col;

        // analytic grid coords, meshgrid 'ij' over linspace(-1,1,128)
        const float fa = -1.0f + (2.0f / 127.0f) * (float)(pidx >> 14);
        const float fb = -1.0f + (2.0f / 127.0f) * (float)((pidx >> 7) & 127);
        const float fc = -1.0f + (2.0f / 127.0f) * (float)(pidx & 127);

        // ---- layer-1 B fragments (registers only) ------------------------
        v16h bk0 = load_B(sX, 0,  0, lane);     // K = 0..31  : latent broadcast
        v16h bk1 = load_B(sX, 32, 0, lane);     // K = 32..63 : latent broadcast
        v16h bk2;
#pragma unroll
        for (int i = 0; i < 16; ++i) bk2[i] = (_Float16)0.0f;
        if (grp == 0) {                          // K = 64..66 : coords, rest 0
            bk2[0] = (_Float16)fa; bk2[1] = (_Float16)fb; bk2[2] = (_Float16)fc;
        }

        // ---- layer 1: 67->128, ReLU -------------------------------------
#pragma unroll
        for (int mt = 0; mt < 8; ++mt) {
            v8f acc = {};
            acc = wmma16(load_A(sW1, mt * 16, 0,  SW1, lane), bk0, acc);
            acc = wmma16(load_A(sW1, mt * 16, 32, SW1, lane), bk1, acc);
            acc = wmma16(load_A(sW1, mt * 16, 64, SW1, lane), bk2, acc);
            const int ch = mt * 16 + grp * 8;
#pragma unroll
            for (int i = 0; i < 8; i += 2) {
                float v0 = fmaxf(acc[i]     + sB1[ch + i],     0.0f);
                float v1 = fmaxf(acc[i + 1] + sB1[ch + i + 1], 0.0f);
                store_pair(actw + col * SACT + ch + i, v0, v1);
            }
        }

        // ---- layer 2: 128->64, ReLU -------------------------------------
        v16h B2[4];
#pragma unroll
        for (int kc = 0; kc < 4; ++kc) B2[kc] = load_B(actw, kc * 32, SACT, lane);
#pragma unroll
        for (int mt = 0; mt < 4; ++mt) {
            v8f acc = {};
#pragma unroll
            for (int kc = 0; kc < 4; ++kc)
                acc = wmma16(load_A(sW2, mt * 16, kc * 32, SW2, lane), B2[kc], acc);
            const int ch = mt * 16 + grp * 8;
#pragma unroll
            for (int i = 0; i < 8; i += 2) {
                float v0 = fmaxf(acc[i]     + sB2[ch + i],     0.0f);
                float v1 = fmaxf(acc[i + 1] + sB2[ch + i + 1], 0.0f);
                store_pair(actw + col * SACT + ch + i, v0, v1);
            }
        }

        // ---- layer 3: 64->32, ReLU --------------------------------------
        v16h B3a = load_B(actw, 0,  SACT, lane);
        v16h B3b = load_B(actw, 32, SACT, lane);
#pragma unroll
        for (int mt = 0; mt < 2; ++mt) {
            v8f acc = {};
            acc = wmma16(load_A(sW3, mt * 16, 0,  SW3, lane), B3a, acc);
            acc = wmma16(load_A(sW3, mt * 16, 32, SW3, lane), B3b, acc);
            const int ch = mt * 16 + grp * 8;
#pragma unroll
            for (int i = 0; i < 8; i += 2) {
                float v0 = fmaxf(acc[i]     + sB3[ch + i],     0.0f);
                float v1 = fmaxf(acc[i + 1] + sB3[ch + i + 1], 0.0f);
                store_pair(actw + col * SACT + ch + i, v0, v1);
            }
        }

        // ---- layer 4: 32->1, sigmoid ------------------------------------
        v16h B4 = load_B(actw, 0, SACT, lane);
        v8f acc = {};
        acc = wmma16(load_A(sW4, 0, 0, SW4, lane), B4, acc);
        // valid output: M=0 row -> acc[0] on lanes 0-15
        float z = acc[0] + sB4;
        float s = 1.0f / (1.0f + __expf(-z));
        if (grp == 0) out[pidx] = s;
    }
}

extern "C" void kernel_launch(void* const* d_in, const int* in_sizes, int n_in,
                              void* d_out, int out_size, void* d_ws, size_t ws_size,
                              hipStream_t stream) {
    (void)in_sizes; (void)n_in; (void)d_ws; (void)ws_size; (void)out_size;
    const float* x  = (const float*)d_in[0];
    const float* W1 = (const float*)d_in[1];
    const float* b1 = (const float*)d_in[2];
    const float* W2 = (const float*)d_in[3];
    const float* b2 = (const float*)d_in[4];
    const float* W3 = (const float*)d_in[5];
    const float* b3 = (const float*)d_in[6];
    const float* W4 = (const float*)d_in[7];
    const float* b4 = (const float*)d_in[8];
    float* out = (float*)d_out;
    sdf_mlp_wmma<<<dim3(BLOCKS), dim3(256), 0, stream>>>(
        x, W1, b1, W2, b2, W3, b3, W4, b4, out);
}
